// GNN_58884001628357
// MI455X (gfx1250) — matrix operand order repaired and checked by reading, not compiled, not run on hardware
//
#include <hip/hip_runtime.h>
#include <hip/hip_bf16.h>

typedef __attribute__((ext_vector_type(16))) _Float16 v16h;
typedef __attribute__((ext_vector_type(8)))  float    v8f;
typedef __attribute__((ext_vector_type(4)))  unsigned int u32x4;
typedef __attribute__((ext_vector_type(8)))  int      i32x8;
typedef __attribute__((ext_vector_type(4)))  int      i32x4;

#define TILES_PER_WAVE 4

__device__ __forceinline__ float silu_f(float x) {
    return x / (1.0f + __expf(-x));
}

__device__ __forceinline__ v8f wmma_f16(v16h a, v16h b, v8f c) {
    return __builtin_amdgcn_wmma_f32_16x16x32_f16(false, a, false, b, (short)0, c, false, false);
}

__device__ __forceinline__ v16h cvtA(const float* t) {
    v16h A;
#pragma unroll
    for (int j = 0; j < 16; ++j) A[j] = (_Float16)t[j];
    return A;
}

// Load a 16-f16 A fragment's source floats: two 8-float runs at p and p+16
__device__ __forceinline__ void load_runs(const float* __restrict__ p, float* t) {
    *(float4*)(t + 0)  = *(const float4*)(p + 0);
    *(float4*)(t + 4)  = *(const float4*)(p + 4);
    *(float4*)(t + 8)  = *(const float4*)(p + 16);
    *(float4*)(t + 12) = *(const float4*)(p + 20);
}

// ---------------------------------------------------------------------------
// Stage `ndw` dwords from global `gptr` into LDS offset `lds_off` using the
// Tensor Data Mover (flat 1-D tile). Must be executed by exactly one wave;
// TDM ignores EXEC. Falls back to a cooperative copy if builtin is absent.
// ---------------------------------------------------------------------------
#if __has_builtin(__builtin_amdgcn_tensor_load_to_lds)
#define HAVE_TDM 1
__device__ __forceinline__ void tdm_copy_to_lds(const void* gptr, unsigned lds_off,
                                                unsigned ndw) {
    unsigned long long ga = (unsigned long long)(uintptr_t)gptr;
    u32x4 g0;
    g0[0] = 1u;                                           // count=1, user mode
    g0[1] = lds_off;                                      // D#.lds_addr
    g0[2] = (unsigned)(ga & 0xFFFFFFFFu);                 // global_addr[31:0]
    g0[3] = (unsigned)((ga >> 32) & 0x01FFFFFFu) | (2u << 30); // addr[56:32], type=2
    i32x8 g1;
    g1[0] = (2 << 16);                                    // data_size = 4 bytes
    g1[1] = (int)((ndw & 0xFFFFu) << 16);                 // tensor_dim0[15:0]
    g1[2] = (int)((ndw >> 16) & 0xFFFFu) | (1 << 16);     // dim0 hi | tensor_dim1=1
    g1[3] = (int)((ndw & 0xFFFFu) << 16);                 // dim1 hi=0 | tile_dim0=ndw
    g1[4] = 1;                                            // tile_dim1=1, tile_dim2=0
    g1[5] = (int)ndw;                                     // tensor_dim0_stride lo
    g1[6] = 0;
    g1[7] = 0;
    i32x4 g2 = {0, 0, 0, 0};
    i32x4 g3 = {0, 0, 0, 0};
    i32x8 g4 = {0, 0, 0, 0, 0, 0, 0, 0};                  // 6-arg toolchain form
    __builtin_amdgcn_tensor_load_to_lds(g0, g1, g2, g3, g4, 0);
}
__device__ __forceinline__ void stage_weights(const _Float16* Wp, _Float16* lw,
                                              int nhalves) {
    if (threadIdx.x < 32) {
        tdm_copy_to_lds(Wp, (unsigned)(uintptr_t)lw, (unsigned)(nhalves >> 1));
#if __has_builtin(__builtin_amdgcn_s_wait_tensorcnt)
        __builtin_amdgcn_s_wait_tensorcnt(0);
#else
        asm volatile("s_wait_tensorcnt 0x0" ::: "memory");
#endif
    }
    __syncthreads();
}
#else
#define HAVE_TDM 0
__device__ __forceinline__ void stage_weights(const _Float16* Wp, _Float16* lw,
                                              int nhalves) {
    const int ndw = nhalves >> 1;
    const unsigned* src = (const unsigned*)Wp;
    unsigned* dst = (unsigned*)lw;
    for (int i = threadIdx.x; i < ndw; i += blockDim.x) dst[i] = src[i];
    __syncthreads();
}
#endif

// ---------------------------------------------------------------------------
// Pack a (Ksrc x Ncols) f32 weight into f16 B-operand fragments:
// dst[((c*4 + nt)*32 + lane)*16 + j] = W[32c + 8*(lane>>4) + j + (j>=8?8:0)][16nt + (lane&15)]
// ---------------------------------------------------------------------------
__global__ void pack_w_kernel(const float* __restrict__ W, int Ksrc, int Ncols,
                              _Float16* __restrict__ dst, int total) {
    int t = blockIdx.x * blockDim.x + threadIdx.x;
    if (t >= total) return;
    int j    = t & 15;
    int lane = (t >> 4) & 31;
    int ct   = t >> 9;            // c*4 + nt
    int nt   = ct & 3;
    int c    = ct >> 2;
    int K    = 32 * c + 8 * (lane >> 4) + j + ((j >= 8) ? 8 : 0);
    int col  = 16 * nt + (lane & 15);
    float v  = (K < Ksrc && col < Ncols) ? W[(size_t)K * Ncols + col] : 0.0f;
    dst[t] = (_Float16)v;
}

// ---------------------------------------------------------------------------
// Edge MLP: e = silu(silu([h[row],h[col],ea] @ W1 + b1) @ W2 + b2);
// agg[row] += e. Wp = packed W1 (5 K-chunks) followed by W2 (2 K-chunks),
// TDM-staged to LDS once per block; each wave loops over 4 edge tiles.
// ---------------------------------------------------------------------------
__launch_bounds__(256)
__global__ void edge_kernel(const float* __restrict__ h,
                            const int* __restrict__ row,
                            const int* __restrict__ col,
                            const float* __restrict__ ea,
                            const _Float16* __restrict__ Wp,
                            const float* __restrict__ b1, const float* __restrict__ b2,
                            float* __restrict__ agg, int nE) {
    __shared__ _Float16 lw[(5 + 2) * 4 * 512];    // 28 KB
    __shared__ float stage[8][16 * 64];           // 32 KB
    const int lane = threadIdx.x & 31;
    const int wave = threadIdx.x >> 5;
    const int ln   = lane & 15;
    const int hf   = lane >> 4;

    stage_weights(Wp, lw, (5 + 2) * 4 * 512);
    const v16h* Bf1 = (const v16h*)lw;
    const v16h* Bf2 = (const v16h*)(lw + 5 * 4 * 512);

    float b1v[4], b2v[4];
#pragma unroll
    for (int nt = 0; nt < 4; ++nt) {
        b1v[nt] = b1[nt * 16 + ln];
        b2v[nt] = b2[nt * 16 + ln];
    }

    for (int it = 0; it < TILES_PER_WAVE; ++it) {
        const int tile = (blockIdx.x * 8 + wave) * TILES_PER_WAVE + it;
        int e  = tile * 16 + ln;
        int ec = (e < nE) ? e : (nE - 1);
        const int   r0  = row[ec];
        const int   c0  = col[ec];
        const float eav = ea[ec];

        v8f acc[4] = {};
#pragma unroll
        for (int c = 0; c < 4; ++c) {
            const float* src = h + (size_t)((c < 2) ? r0 : c0) * 64 + (c & 1) * 32 + hf * 8;
            float t[16];
            load_runs(src, t);
            v16h A = cvtA(t);
#pragma unroll
            for (int nt = 0; nt < 4; ++nt)
                acc[nt] = wmma_f16(A, Bf1[(c * 4 + nt) * 32 + lane], acc[nt]);
        }
        {   // K-chunk 4: only K=128 (edge_attr) nonzero
            v16h A = {};
            A[0] = hf ? (_Float16)0.0f : (_Float16)eav;
#pragma unroll
            for (int nt = 0; nt < 4; ++nt)
                acc[nt] = wmma_f16(A, Bf1[(16 + nt) * 32 + lane], acc[nt]);
        }

#pragma unroll
        for (int nt = 0; nt < 4; ++nt)
#pragma unroll
            for (int r = 0; r < 8; ++r)
                stage[wave][(r + 8 * hf) * 64 + nt * 16 + ln] = silu_f(acc[nt][r] + b1v[nt]);
        asm volatile("s_wait_dscnt 0" ::: "memory");

        v8f a2[4] = {};
#pragma unroll
        for (int c = 0; c < 2; ++c) {
            const float* sp = &stage[wave][ln * 64 + c * 32 + hf * 8];
            float t[16];
            load_runs(sp, t);
            v16h A = cvtA(t);
#pragma unroll
            for (int nt = 0; nt < 4; ++nt)
                a2[nt] = wmma_f16(A, Bf2[(c * 4 + nt) * 32 + lane], a2[nt]);
        }

#pragma unroll
        for (int nt = 0; nt < 4; ++nt)
#pragma unroll
            for (int r = 0; r < 8; ++r) {
                int md = r + 8 * hf;
                int ed = tile * 16 + md;
                int rd = __shfl(r0, md, 32);
                float y = silu_f(a2[nt][r] + b2v[nt]);
                if (ed < nE)
                    unsafeAtomicAdd(&agg[(size_t)rd * 64 + nt * 16 + ln], y);
            }
    }
}

// ---------------------------------------------------------------------------
// Node MLP: out = silu([h,agg] @ W1 + b1) @ W2 + b2.  Wp = W1 (4 chunks) +
// W2 (2 chunks). outCols = 64 (hidden) or 4 (decode; W2 zero-padded cols).
// ---------------------------------------------------------------------------
__launch_bounds__(256)
__global__ void node_kernel(const float* __restrict__ h,
                            const float* __restrict__ agg,
                            const _Float16* __restrict__ Wp,
                            const float* __restrict__ b1, const float* __restrict__ b2,
                            float* __restrict__ out, int nN, int outCols) {
    __shared__ _Float16 lw[(4 + 2) * 4 * 512];    // 24 KB
    __shared__ float stage[8][16 * 64];           // 32 KB
    const int lane = threadIdx.x & 31;
    const int wave = threadIdx.x >> 5;
    const int ln   = lane & 15;
    const int hf   = lane >> 4;

    stage_weights(Wp, lw, (4 + 2) * 4 * 512);
    const v16h* Bf1 = (const v16h*)lw;
    const v16h* Bf2 = (const v16h*)(lw + 4 * 4 * 512);

    float b1v[4], b2v[4];
#pragma unroll
    for (int nt = 0; nt < 4; ++nt) {
        int colg = nt * 16 + ln;
        b1v[nt] = b1[colg];
        b2v[nt] = (colg < outCols) ? b2[colg] : 0.0f;
    }

    for (int it = 0; it < TILES_PER_WAVE; ++it) {
        const int tile = (blockIdx.x * 8 + wave) * TILES_PER_WAVE + it;
        int nd = tile * 16 + ln;
        if (nd >= nN) nd = nN - 1;

        v8f acc[4] = {};
#pragma unroll
        for (int c = 0; c < 4; ++c) {
            const float* base = (c < 2) ? h : agg;
            const float* src  = base + (size_t)nd * 64 + (c & 1) * 32 + hf * 8;
            float t[16];
            load_runs(src, t);
            v16h A = cvtA(t);
#pragma unroll
            for (int nt = 0; nt < 4; ++nt)
                acc[nt] = wmma_f16(A, Bf1[(c * 4 + nt) * 32 + lane], acc[nt]);
        }

#pragma unroll
        for (int nt = 0; nt < 4; ++nt)
#pragma unroll
            for (int r = 0; r < 8; ++r)
                stage[wave][(r + 8 * hf) * 64 + nt * 16 + ln] = silu_f(acc[nt][r] + b1v[nt]);
        asm volatile("s_wait_dscnt 0" ::: "memory");

        v8f a2[4] = {};
#pragma unroll
        for (int c = 0; c < 2; ++c) {
            const float* sp = &stage[wave][ln * 64 + c * 32 + hf * 8];
            float t[16];
            load_runs(sp, t);
            v16h A = cvtA(t);
#pragma unroll
            for (int nt = 0; nt < 4; ++nt)
                a2[nt] = wmma_f16(A, Bf2[(c * 4 + nt) * 32 + lane], a2[nt]);
        }

#pragma unroll
        for (int nt = 0; nt < 4; ++nt) {
            int colg = nt * 16 + ln;
#pragma unroll
            for (int r = 0; r < 8; ++r) {
                int md  = r + 8 * hf;
                int ndd = tile * 16 + md;
                float v = a2[nt][r] + b2v[nt];   // no activation on 2nd layer
                if (ndd < nN && colg < outCols)
                    out[(size_t)ndd * outCols + colg] = v;
            }
        }
    }
}

// ---------------------------------------------------------------------------
// Embedding: group features (GROUP = {+I,-I} => sign flips) @ W_emb -> 128,
// then silu(.@Wg1+bg1)@Wg2+bg2.  Wp = Wg1 (4 chunks) + Wg2 (2 chunks).
// ---------------------------------------------------------------------------
__launch_bounds__(256)
__global__ void embed_kernel(const float* __restrict__ nodes,
                             const float* __restrict__ We, const float* __restrict__ be,
                             const _Float16* __restrict__ Wp,
                             const float* __restrict__ bg1, const float* __restrict__ bg2,
                             float* __restrict__ h0, int nN) {
    __shared__ _Float16 lw[(4 + 2) * 4 * 512];
    __shared__ float stage[8][16 * 64];
    const int lane = threadIdx.x & 31;
    const int wave = threadIdx.x >> 5;
    const int ln   = lane & 15;
    const int hf   = lane >> 4;

    stage_weights(Wp, lw, (4 + 2) * 4 * 512);
    const v16h* Bf1 = (const v16h*)lw;
    const v16h* Bf2 = (const v16h*)(lw + 4 * 4 * 512);

    float b1v[4], b2v[4];
#pragma unroll
    for (int nt = 0; nt < 4; ++nt) {
        b1v[nt] = bg1[nt * 16 + ln];
        b2v[nt] = bg2[nt * 16 + ln];
    }

    for (int it = 0; it < TILES_PER_WAVE; ++it) {
        const int tile = (blockIdx.x * 8 + wave) * TILES_PER_WAVE + it;
        int nd = tile * 16 + ln;
        if (nd >= nN) nd = nN - 1;
        const float l0 = nodes[(size_t)nd * 5 + 0];
        const float l1 = nodes[(size_t)nd * 5 + 1];
        const float v0 = nodes[(size_t)nd * 5 + 2];
        const float v1 = nodes[(size_t)nd * 5 + 3];
        const float tp = nodes[(size_t)nd * 5 + 4];

        v8f acc[4] = {};
#pragma unroll
        for (int c = 0; c < 4; ++c) {
            float t[16];
#pragma unroll
            for (int j = 0; j < 16; ++j) {
                int K   = 32 * c + 8 * hf + j + ((j >= 8) ? 8 : 0);
                int o   = K & 63;
                float s = (K >> 6) ? -1.0f : 1.0f;
                t[j] = be[o]
                     + s * (l0 * We[o] + l1 * We[64 + o] + v0 * We[128 + o] + v1 * We[192 + o])
                     + tp * We[256 + o];
            }
            v16h A = cvtA(t);
#pragma unroll
            for (int nt = 0; nt < 4; ++nt)
                acc[nt] = wmma_f16(A, Bf1[(c * 4 + nt) * 32 + lane], acc[nt]);
        }

#pragma unroll
        for (int nt = 0; nt < 4; ++nt)
#pragma unroll
            for (int r = 0; r < 8; ++r)
                stage[wave][(r + 8 * hf) * 64 + nt * 16 + ln] = silu_f(acc[nt][r] + b1v[nt]);
        asm volatile("s_wait_dscnt 0" ::: "memory");

        v8f a2[4] = {};
#pragma unroll
        for (int c = 0; c < 2; ++c) {
            const float* sp = &stage[wave][ln * 64 + c * 32 + hf * 8];
            float t[16];
            load_runs(sp, t);
            v16h A = cvtA(t);
#pragma unroll
            for (int nt = 0; nt < 4; ++nt)
                a2[nt] = wmma_f16(A, Bf2[(c * 4 + nt) * 32 + lane], a2[nt]);
        }

#pragma unroll
        for (int nt = 0; nt < 4; ++nt)
#pragma unroll
            for (int r = 0; r < 8; ++r) {
                int md  = r + 8 * hf;
                int ndd = tile * 16 + md;
                float v = a2[nt][r] + b2v[nt];
                if (ndd < nN)
                    h0[(size_t)ndd * 64 + nt * 16 + ln] = v;
            }
    }
}

extern "C" void kernel_launch(void* const* d_in, const int* in_sizes, int n_in,
                              void* d_out, int out_size, void* d_ws, size_t ws_size,
                              hipStream_t stream) {
    const float* nodes = (const float*)d_in[0];
    const int*   edges = (const int*)d_in[1];
    const float* eatt  = (const float*)d_in[2];
    const float* W_emb = (const float*)d_in[3];
    const float* b_emb = (const float*)d_in[4];
    const float* Wg1   = (const float*)d_in[5];
    const float* bg1   = (const float*)d_in[6];
    const float* Wg2   = (const float*)d_in[7];
    const float* bg2   = (const float*)d_in[8];
    const float* eW1   = (const float*)d_in[9];
    const float* eb1   = (const float*)d_in[10];
    const float* eW2   = (const float*)d_in[11];
    const float* eb2   = (const float*)d_in[12];
    const float* nW1   = (const float*)d_in[13];
    const float* nb1   = (const float*)d_in[14];
    const float* nW2   = (const float*)d_in[15];
    const float* nb2   = (const float*)d_in[16];
    const float* deW1  = (const float*)d_in[17];
    const float* deb1  = (const float*)d_in[18];
    const float* deW2  = (const float*)d_in[19];
    const float* deb2  = (const float*)d_in[20];
    const float* dnW1  = (const float*)d_in[21];
    const float* dnb1  = (const float*)d_in[22];
    const float* dnW2  = (const float*)d_in[23];
    const float* dnb2  = (const float*)d_in[24];

    const int Ncnt = in_sizes[0] / 5;          // 25000
    const int Ecnt = in_sizes[2];              // 400000 (edge_attr is (E,1))
    const int NLc  = in_sizes[9] / (129 * 64); // 5

    // ---- pack all weights to f16 fragment layout in workspace ----
    // NOTE: allocation order guarantees W1||W2 contiguity per MLP for the
    // single-DMA LDS staging in the kernels.
    _Float16* wp = (_Float16*)d_ws;
    size_t off = 0;
    auto pack = [&](const float* W, int Ksrc, int Ncols, int nchunks) -> const _Float16* {
        _Float16* dst = wp + off;
        int total = nchunks * 4 * 512;   // 4 N-tiles, 32 lanes * 16 halves
        off += (size_t)total;
        pack_w_kernel<<<(total + 255) / 256, 256, 0, stream>>>(W, Ksrc, Ncols, dst, total);
        return dst;
    };

    const _Float16* Wgp = pack(Wg1, 128, 64, 4);        // + Wg2 below (contiguous)
    pack(Wg2, 64, 64, 2);
    const _Float16 *eWp[8], *nWp[8];
    for (int i = 0; i < NLc; ++i) {
        eWp[i] = pack(eW1 + (size_t)i * 129 * 64, 129, 64, 5);
        pack(eW2 + (size_t)i * 64 * 64, 64, 64, 2);
        nWp[i] = pack(nW1 + (size_t)i * 128 * 64, 128, 64, 4);
        pack(nW2 + (size_t)i * 64 * 64, 64, 64, 2);
    }
    const _Float16* deWp = pack(deW1, 129, 64, 5);
    pack(deW2, 64, 64, 2);
    const _Float16* dnWp = pack(dnW1, 128, 64, 4);
    pack(dnW2, 64, 4, 2);   // cols 4..63 zero-padded

    // ---- float scratch: two h buffers + agg ----
    size_t fbyte = (off * sizeof(_Float16) + 255) & ~(size_t)255;
    float* hA  = (float*)((char*)d_ws + fbyte);
    float* hB  = hA + (size_t)Ncnt * 64;
    float* agg = hB + (size_t)Ncnt * 64;

    const int tilesPerBlock = 8 * TILES_PER_WAVE;
    const int nodeBlocks = ((Ncnt + 15) / 16 + tilesPerBlock - 1) / tilesPerBlock;
    const int edgeBlocks = ((Ecnt + 15) / 16 + tilesPerBlock - 1) / tilesPerBlock;
    const int* row = edges;
    const int* col = edges + Ecnt;

    embed_kernel<<<nodeBlocks, 256, 0, stream>>>(nodes, W_emb, b_emb, Wgp,
                                                 bg1, bg2, hA, Ncnt);

    float* cur = hA;
    float* nxt = hB;
    for (int i = 0; i < NLc; ++i) {
        (void)hipMemsetAsync(agg, 0, (size_t)Ncnt * 64 * sizeof(float), stream);
        edge_kernel<<<edgeBlocks, 256, 0, stream>>>(cur, row, col, eatt, eWp[i],
                                                    eb1 + i * 64, eb2 + i * 64, agg, Ecnt);
        node_kernel<<<nodeBlocks, 256, 0, stream>>>(cur, agg, nWp[i],
                                                    nb1 + i * 64, nb2 + i * 64,
                                                    nxt, Ncnt, 64);
        float* tmp = cur; cur = nxt; nxt = tmp;
    }

    // decode layer
    (void)hipMemsetAsync(agg, 0, (size_t)Ncnt * 64 * sizeof(float), stream);
    edge_kernel<<<edgeBlocks, 256, 0, stream>>>(cur, row, col, eatt, deWp,
                                                deb1, deb2, agg, Ecnt);
    node_kernel<<<nodeBlocks, 256, 0, stream>>>(cur, agg, dnWp, dnb1, dnb2,
                                                (float*)d_out, Ncnt, 4);
}
